// DLRM_69355131896386
// MI455X (gfx1250) — matrix operand order repaired and checked by reading, not compiled, not run on hardware
//
#include <hip/hip_runtime.h>
#include <stdint.h>

// ---------------- problem constants ----------------
#define B_SAMPLES 16384
#define F_FEAT    26
#define V_VOCAB   100000
#define D_EMB     64
#define H1_DIM    512
#define H2_DIM    256
#define N_INTER   325              // F*(F-1)/2
#define RAW_IN    1989             // F*D + N_INTER
#define K1_PAD    2048             // padded K for GEMM1 (mult of 32)

typedef __attribute__((ext_vector_type(16))) __bf16 v16bf;
typedef __attribute__((ext_vector_type(8)))  float  v8f;

__device__ __forceinline__ unsigned short f32_to_bf16_rne(float f) {
    unsigned int u = __float_as_uint(f);
    u += 0x7FFFu + ((u >> 16) & 1u);       // round-to-nearest-even
    return (unsigned short)(u >> 16);
}
__device__ __forceinline__ float bf16_to_f32(unsigned short h) {
    return __uint_as_float(((unsigned int)h) << 16);
}

// ---- CDNA5 async DMA: per-lane global -> LDS, 16 B, tracked by ASYNCcnt ----
__device__ __forceinline__ void async_copy_b128(void* lds_dst, const void* gsrc) {
    unsigned lds_addr = (unsigned)(size_t)lds_dst;
    unsigned long long gaddr = (unsigned long long)(size_t)gsrc;
    asm volatile("global_load_async_to_lds_b128 %0, %1, off"
                 :: "v"(lds_addr), "v"(gaddr)
                 : "memory");
}
__device__ __forceinline__ void wait_async0() {
    asm volatile("s_wait_asynccnt 0x0" ::: "memory");
}

// -------- weight conversion: f32 [rows_src x cols] -> bf16 TRANSPOSED [cols x rows_dst] --------
__global__ void convert_transpose_bf16(const float* __restrict__ src,
                                       unsigned short* __restrict__ dst,
                                       int rows_src, int rows_dst, int cols) {
    size_t g = (size_t)blockIdx.x * blockDim.x + threadIdx.x;
    size_t total = (size_t)cols * rows_dst;
    if (g >= total) return;
    int n = (int)(g / (size_t)rows_dst);
    int k = (int)(g % (size_t)rows_dst);
    float v = (k < rows_src) ? src[(size_t)k * cols + n] : 0.0f;
    dst[g] = f32_to_bf16_rne(v);
}

// ---------------- gather + pairwise interactions -> joint (bf16) ----------------
__global__ void __launch_bounds__(256)
embed_interact(const int* __restrict__ idx, const float* __restrict__ tables,
               unsigned short* __restrict__ joint) {
    __shared__ int   sidx[F_FEAT];
    __shared__ float semb[F_FEAT * D_EMB];           // 6.5 KB
    const int b = blockIdx.x;
    const int t = threadIdx.x;

    if (t < F_FEAT) sidx[t] = idx[b * F_FEAT + t];
    __syncthreads();

    unsigned short* row = joint + (size_t)b * K1_PAD;

    for (int e = t; e < F_FEAT * D_EMB; e += 256) {
        int f = e >> 6, d = e & 63;
        float v = tables[((size_t)f * V_VOCAB + (size_t)sidx[f]) * D_EMB + d];
        semb[e] = v;
        row[e]  = f32_to_bf16_rne(v);
    }
    __syncthreads();

    for (int p = t; p < N_INTER; p += 256) {
        int i = 0, rem = p, cnt = F_FEAT - 1;
        while (rem >= cnt) { rem -= cnt; ++i; --cnt; }
        int j = i + 1 + rem;
        const float4* a4 = (const float4*)(semb + i * D_EMB);
        const float4* c4 = (const float4*)(semb + j * D_EMB);
        float acc = 0.f;
#pragma unroll
        for (int q = 0; q < D_EMB / 4; ++q) {
            float4 av = a4[q], cv = c4[q];
            acc += av.x * cv.x + av.y * cv.y + av.z * cv.z + av.w * cv.w;
        }
        row[F_FEAT * D_EMB + p] = f32_to_bf16_rne(acc);
    }

    for (int c = RAW_IN + t; c < K1_PAD; c += 256) row[c] = 0;
}

// ---------------- bf16 WMMA GEMM: C = act(A @ Wt^T + bias) ----------------
// A : M x K row-major bf16;  Wt: N x K row-major bf16 (pre-transposed weights)
// Block tile 128x128, 8 waves; wave computes 32x64 (8 WMMA tiles / K-step).
// GLOBAL_LOAD_ASYNC_TO_LDS_B128 staging (ASYNCcnt), double-buffered LDS with
// the K-loop unrolled x2 so buffer indices are compile-time constants.
// Requires kSteps = K/32 even (holds for K = 2048 and 512).
template <int RELU>
__global__ void __launch_bounds__(256)
gemm_bf16_bias(const unsigned short* __restrict__ A,
               const unsigned short* __restrict__ Wt,
               const float* __restrict__ bias,
               unsigned short* __restrict__ C,
               int M, int N, int K) {
    __shared__ __align__(16) unsigned short As[2][128 * 40];   // 20 KB
    __shared__ __align__(16) unsigned short Bs[2][128 * 40];   // 20 KB

    const int t    = threadIdx.x;
    const int lane = t & 31;
    const int wave = t >> 5;
    const int wr   = wave >> 1;         // 0..3 : 32-row band
    const int wc   = wave & 1;          // 0..1 : 64-col band
    const int m0   = blockIdx.y * 128;
    const int n0   = blockIdx.x * 128;

    const int lanehalf = lane >> 4;     // K half (per ISA 16-bit layouts)
    const int lanem    = lane & 15;

    // staging: both tiles are 128 rows x 32 k; 2 threads/row, 16 bf16 (32 B) each
    const int srow = t >> 1;
    const int skg  = (t & 1) * 16;
    const int soff = srow * 40 + skg;

    const unsigned short* Aptr = A  + (size_t)(m0 + srow) * K + skg;
    const unsigned short* Bptr = Wt + (size_t)(n0 + srow) * K + skg;

    v8f acc[2][4] = {};
    const int kSteps = K >> 5;

    auto stage = [&](int bufc, int step) {
        const unsigned short* ap = Aptr + (size_t)step * 32;
        const unsigned short* bp = Bptr + (size_t)step * 32;
        async_copy_b128(&As[bufc][soff],     ap);
        async_copy_b128(&As[bufc][soff + 8], ap + 8);
        async_copy_b128(&Bs[bufc][soff],     bp);
        async_copy_b128(&Bs[bufc][soff + 8], bp + 8);
    };
    auto l2_prefetch = [&](int step) {   // speculative: safe past the end
        __builtin_prefetch(Aptr + (size_t)step * 32, 0, 1);
        __builtin_prefetch(Bptr + (size_t)step * 32, 0, 1);
    };
    auto compute = [&](int bufc) {
        v16bf af[2], bfr[4];
#pragma unroll
        for (int tm = 0; tm < 2; ++tm) {
            int mrow = wr * 32 + tm * 16 + lanem;
            ((uint4*)&af[tm])[0] = *(const uint4*)(&As[bufc][mrow * 40 + lanehalf * 8]);
            ((uint4*)&af[tm])[1] = *(const uint4*)(&As[bufc][mrow * 40 + lanehalf * 8 + 16]);
        }
#pragma unroll
        for (int tn = 0; tn < 4; ++tn) {
            int ncol = wc * 64 + tn * 16 + lanem;
            ((uint4*)&bfr[tn])[0] = *(const uint4*)(&Bs[bufc][ncol * 40 + lanehalf * 8]);
            ((uint4*)&bfr[tn])[1] = *(const uint4*)(&Bs[bufc][ncol * 40 + lanehalf * 8 + 16]);
        }
#pragma unroll
        for (int tm = 0; tm < 2; ++tm)
#pragma unroll
            for (int tn = 0; tn < 4; ++tn)
                acc[tm][tn] = __builtin_amdgcn_wmma_f32_16x16x32_bf16(
                    false, af[tm], false, bfr[tn],
                    (short)0, acc[tm][tn], false, false);
    };

    // prologue: async-stage tile 0 into buffer 0
    stage(0, 0);

    // kSteps even: unrolled x2 with constant buffer indices
    for (int s = 0; s < kSteps; s += 2) {
        wait_async0();          // buffer 0's async writes landed (this wave)
        __syncthreads();        // publish buffer 0
        stage(1, s + 1);        // always valid: s+1 <= kSteps-1
        l2_prefetch(s + 2);
        compute(0);

        wait_async0();          // buffer 1 ready
        __syncthreads();
        if (s + 2 < kSteps) stage(0, s + 2);
        l2_prefetch(s + 3);
        compute(1);
    }

    // epilogue: bias + activation, store bf16
#pragma unroll
    for (int tm = 0; tm < 2; ++tm) {
#pragma unroll
        for (int tn = 0; tn < 4; ++tn) {
            int col = n0 + wc * 64 + tn * 16 + lanem;
            float bv = bias[col];
            int rbase = m0 + wr * 32 + tm * 16 + lanehalf * 8;
#pragma unroll
            for (int g = 0; g < 8; ++g) {
                float v = acc[tm][tn][g] + bv;
                if (RELU) v = v > 0.f ? v : 0.f;
                C[(size_t)(rbase + g) * N + col] = f32_to_bf16_rne(v);
            }
        }
    }
}

// ---------------- final layer: out[b] = h2[b,:] . W3 + b3 ----------------
__global__ void __launch_bounds__(256)
final_layer(const unsigned short* __restrict__ h2,
            const float* __restrict__ W3, const float* __restrict__ b3,
            float* __restrict__ out) {
    __shared__ float sw[H2_DIM];
    int t = threadIdx.x;
    sw[t] = W3[t];
    __syncthreads();
    int b = blockIdx.x * 256 + t;
    const unsigned short* row = h2 + (size_t)b * H2_DIM;
    float acc = b3[0];
#pragma unroll 8
    for (int j = 0; j < H2_DIM; ++j) acc += bf16_to_f32(row[j]) * sw[j];
    out[b] = acc;
}

// ---------------- launch ----------------
extern "C" void kernel_launch(void* const* d_in, const int* in_sizes, int n_in,
                              void* d_out, int out_size, void* d_ws, size_t ws_size,
                              hipStream_t stream) {
    const int*   idx    = (const int*)d_in[0];
    const float* tables = (const float*)d_in[1];
    const float* W1     = (const float*)d_in[2];
    const float* b1     = (const float*)d_in[3];
    const float* W2     = (const float*)d_in[4];
    const float* b2     = (const float*)d_in[5];
    const float* W3     = (const float*)d_in[6];
    const float* b3     = (const float*)d_in[7];
    float* out = (float*)d_out;

    // workspace layout (bytes)
    char* ws = (char*)d_ws;
    unsigned short* joint = (unsigned short*)(ws);                                   // 64 MB
    unsigned short* h1    = (unsigned short*)(ws + 67108864);                        // 16 MB
    unsigned short* h2    = (unsigned short*)(ws + 67108864 + 16777216);             //  8 MB
    unsigned short* W1t   = (unsigned short*)(ws + 67108864 + 16777216 + 8388608);   //  2 MB
    unsigned short* W2t   = (unsigned short*)(ws + 67108864 + 16777216 + 8388608 + 2097152);

    convert_transpose_bf16<<<(H1_DIM * K1_PAD) / 256, 256, 0, stream>>>(
        W1, W1t, RAW_IN, K1_PAD, H1_DIM);
    convert_transpose_bf16<<<(H2_DIM * H1_DIM) / 256, 256, 0, stream>>>(
        W2, W2t, H1_DIM, H1_DIM, H2_DIM);

    embed_interact<<<B_SAMPLES, 256, 0, stream>>>(idx, tables, joint);

    gemm_bf16_bias<1><<<dim3(H1_DIM / 128, B_SAMPLES / 128), 256, 0, stream>>>(
        joint, W1t, b1, h1, B_SAMPLES, H1_DIM, K1_PAD);

    gemm_bf16_bias<1><<<dim3(H2_DIM / 128, B_SAMPLES / 128), 256, 0, stream>>>(
        h1, W2t, b2, h2, B_SAMPLES, H2_DIM, H1_DIM);

    final_layer<<<B_SAMPLES / 256, 256, 0, stream>>>(h2, W3, b3, out);
}
